// LocalFeatureAggregation_70411693851257
// MI455X (gfx1250) — compile-verified, matrix operand order checked by reading, not run on hardware
//
#include <hip/hip_runtime.h>

#define BATCH   2
#define NPTS    8192
#define KNN     16
#define CIN     64
#define COUT    128
#define HPRE    64
#define EIN     132
#define EINP    160      // padded K for layer 1 (5 x 32)
#define ES      168      // edge LDS row stride in halfs (16B aligned: 336B)
#define H1S     136      // h1 LDS row stride in halfs (272B, 16B aligned)
#define TILE_PTS 16
#define EPSBN   1e-5f
#define SLOPE   0.2f

typedef __attribute__((ext_vector_type(16))) _Float16 v16h;
typedef __attribute__((ext_vector_type(8)))  float    v8f;

// LeakyReLU with slope<1 as a single max: lrelu(y) = max(y, slope*y).
__device__ __forceinline__ float lrelu(float y) { return fmaxf(y, SLOPE * y); }

// Load one 16x32 f16 WMMA operand tile row for this lane.
// Per CDNA5 ISA 16-bit A layout: lanes 0-15 hold K[0..7] and K[16..23] of row
// (lane&15); lanes 16-31 hold K[8..15] and K[24..31].  B mirrors this with
// n = lane&15 selecting a row of W stored row-major [N][K].
__device__ __forceinline__ v16h load_tile16(const _Float16* rowk, int lane) {
    const int off = (lane >> 4) << 3;                 // 0 or 8 halfs
    union { uint4 u[2]; v16h h; } r;
    r.u[0] = *reinterpret_cast<const uint4*>(rowk + off);
    r.u[1] = *reinterpret_cast<const uint4*>(rowk + 16 + off);
    return r.h;
}

// ---------------------------------------------------------------- kNN -------
__global__ void __launch_bounds__(256)
knn_kernel(const float* __restrict__ xyz, int* __restrict__ idx) {
    const int b  = blockIdx.y;
    const int pt = blockIdx.x * 256 + threadIdx.x;
    const float* xb = xyz + (size_t)b * NPTS * 3;
    const float qx = xb[pt * 3 + 0];
    const float qy = xb[pt * 3 + 1];
    const float qz = xb[pt * 3 + 2];

    float bd[KNN]; int bi[KNN];
#pragma unroll
    for (int s = 0; s < KNN; ++s) { bd[s] = 3.4e38f; bi[s] = 0; }

    __shared__ float sx[256], sy[256], sz[256];
    for (int j0 = 0; j0 < NPTS; j0 += 256) {
        __syncthreads();
        sx[threadIdx.x] = xb[(j0 + threadIdx.x) * 3 + 0];
        sy[threadIdx.x] = xb[(j0 + threadIdx.x) * 3 + 1];
        sz[threadIdx.x] = xb[(j0 + threadIdx.x) * 3 + 2];
        __syncthreads();
        for (int jj = 0; jj < 256; ++jj) {
            const int j = j0 + jj;
            const float dx = sx[jj] - qx, dy = sy[jj] - qy, dz = sz[jj] - qz;
            const float d2 = fmaf(dx, dx, fmaf(dy, dy, dz * dz));
            if (j != pt && d2 < bd[KNN - 1]) {        // rare-branch guard
                float dc = d2; int ic = j;
#pragma unroll
                for (int s = 0; s < KNN; ++s) {       // unrolled insertion chain
                    if (dc < bd[s]) {
                        float td = bd[s]; int ti = bi[s];
                        bd[s] = dc; bi[s] = ic; dc = td; ic = ti;
                    }
                }
            }
        }
    }
    int* op = idx + ((size_t)b * NPTS + pt) * KNN;
#pragma unroll
    for (int s = 0; s < KNN; ++s) op[s] = bi[s];
}

// ------------------------------------------ weight convert + BN folding -----
__global__ void __launch_bounds__(256)
prep_kernel(const float* __restrict__ w_e1, const float* __restrict__ bn_e1,
            const float* __restrict__ w_e2, const float* __restrict__ bn_e2,
            _Float16* __restrict__ w1h, _Float16* __restrict__ w2h,
            float* __restrict__ e1s, float* __restrict__ e1b,
            float* __restrict__ e2s, float* __restrict__ e2b) {
    const int t = threadIdx.x;
    for (int i = t; i < COUT * EINP; i += 256) {
        const int r = i / EINP, c = i % EINP;
        const float v = (c < EIN) ? w_e1[r * EIN + c] : 0.f;  // zero-pad K
        w1h[i] = (_Float16)v;
    }
    for (int i = t; i < COUT * COUT; i += 256) w2h[i] = (_Float16)w_e2[i];
    if (t < COUT) {
        float g = bn_e1[t], be = bn_e1[COUT + t];
        float m = bn_e1[2 * COUT + t], v = bn_e1[3 * COUT + t];
        float s = g * rsqrtf(v + EPSBN);
        e1s[t] = s; e1b[t] = be - m * s;
        g = bn_e2[t]; be = bn_e2[COUT + t];
        m = bn_e2[2 * COUT + t]; v = bn_e2[3 * COUT + t];
        s = g * rsqrtf(v + EPSBN);
        e2s[t] = s; e2b[t] = be - m * s;
    }
}

// --------------------------------------- pre-MLP (f16 out) + shortcut ------
__global__ void __launch_bounds__(256)
pre_kernel(const float* __restrict__ features, const float* __restrict__ w_pre,
           const float* __restrict__ bn_pre,  const float* __restrict__ w_sc,
           const float* __restrict__ bn_sc,
           _Float16* __restrict__ fh, float* __restrict__ sc) {
    const int b = blockIdx.y;
    const int g = threadIdx.x >> 6;                  // 4 points per block
    const int n = blockIdx.x * 4 + g;
    const int t = threadIdx.x & 63;                  // channel / hidden index

    __shared__ float fbuf[4][CIN];
    fbuf[g][t] = features[(size_t)b * CIN * NPTS + (size_t)t * NPTS + n];
    __syncthreads();
    const float* fb = fbuf[g];

    // pre-MLP hidden unit t
    float acc = 0.f;
    const float* wr = w_pre + t * CIN;
#pragma unroll 8
    for (int c = 0; c < CIN; ++c) acc = fmaf(wr[c], fb[c], acc);
    {
        const float gm = bn_pre[t], be = bn_pre[HPRE + t];
        const float mn = bn_pre[2 * HPRE + t], vr = bn_pre[3 * HPRE + t];
        const float s = gm * rsqrtf(vr + EPSBN);
        const float y = lrelu((acc - mn) * s + be);
        fh[((size_t)b * NPTS + n) * HPRE + t] = (_Float16)y;
    }
    // shortcut channels t and t+64 (BN only; LReLU applied after add)
#pragma unroll
    for (int h = 0; h < 2; ++h) {
        const int c = t + 64 * h;
        float a2 = 0.f;
        const float* wr2 = w_sc + c * CIN;
#pragma unroll 8
        for (int k = 0; k < CIN; ++k) a2 = fmaf(wr2[k], fb[k], a2);
        const float gm = bn_sc[c], be = bn_sc[COUT + c];
        const float mn = bn_sc[2 * COUT + c], vr = bn_sc[3 * COUT + c];
        const float s = gm * rsqrtf(vr + EPSBN);
        sc[((size_t)b * NPTS + n) * COUT + c] = (a2 - mn) * s + be;
    }
}

// ------------------------------------------- fused edge MLPs (WMMA core) ----
__global__ void __launch_bounds__(256)
edge_kernel(const float* __restrict__ xyz, const int* __restrict__ idx,
            const _Float16* __restrict__ fh, const float* __restrict__ sc,
            const _Float16* __restrict__ w1h, const _Float16* __restrict__ w2h,
            const float* __restrict__ e1s, const float* __restrict__ e1b,
            const float* __restrict__ e2s, const float* __restrict__ e2b,
            float* __restrict__ out) {
    extern __shared__ _Float16 smem[];
    _Float16* edge = smem;                           // [256][ES]
    _Float16* h1   = smem + 256 * ES;                // [256][H1S]

    const int b    = blockIdx.y;
    const int pt0  = blockIdx.x * TILE_PTS;
    const int tid  = threadIdx.x;
    const int lane = tid & 31;
    const int wv   = tid >> 5;                       // wave = N-tile id, 0..7

    __builtin_prefetch(w1h, 0, 1);
    __builtin_prefetch(w2h, 0, 1);

    // Each wave keeps its B operands (one 16-channel slab of W1^T / W2^T)
    // resident in VGPRs for the whole workgroup.
    const int crow = wv * 16 + (lane & 15);          // output channel per lane
    v16h B1[5], B2[4];
#pragma unroll
    for (int ks = 0; ks < 5; ++ks)
        B1[ks] = load_tile16(w1h + (size_t)crow * EINP + ks * 32, lane);
#pragma unroll
    for (int ks = 0; ks < 4; ++ks)
        B2[ks] = load_tile16(w2h + (size_t)crow * COUT + ks * 32, lane);
    const float s1 = e1s[crow], o1 = e1b[crow];
    const float s2 = e2s[crow], o2 = e2b[crow];

    // ---- build the 256 x 160 f16 edge-feature matrix in LDS ----
    {
        const int lp = tid >> 4;                     // local point 0..15
        const int kk = tid & 15;                     // neighbor slot
        const int pt = pt0 + lp;
        const int nb = idx[((size_t)b * NPTS + pt) * KNN + kk];
        const float* xb = xyz + (size_t)b * NPTS * 3;
        const float cx = xb[pt * 3 + 0], cy = xb[pt * 3 + 1], cz = xb[pt * 3 + 2];
        const float rx = xb[nb * 3 + 0] - cx;
        const float ry = xb[nb * 3 + 1] - cy;
        const float rz = xb[nb * 3 + 2] - cz;
        const float dd = sqrtf(fmaf(rx, rx, fmaf(ry, ry, rz * rz)));

        const uint4* cfv = reinterpret_cast<const uint4*>(fh + ((size_t)b * NPTS + pt) * HPRE);
        const uint4* nfv = reinterpret_cast<const uint4*>(fh + ((size_t)b * NPTS + nb) * HPRE);
        _Float16* er = edge + (size_t)tid * ES;      // edge row = lp*16 + kk
#pragma unroll
        for (int q = 0; q < 8; ++q) {                // 8 halfs per chunk
            union { uint4 u; _Float16 h[8]; } cu, nu, co, no;
            cu.u = cfv[q]; nu.u = nfv[q];
#pragma unroll
            for (int i = 0; i < 8; ++i) {
                co.h[i] = cu.h[i];                                       // ctr_f
                no.h[i] = (_Float16)((float)nu.h[i] - (float)cu.h[i]);   // nbr-ctr
            }
            *reinterpret_cast<uint4*>(er + q * 8)        = co.u;
            *reinterpret_cast<uint4*>(er + HPRE + q * 8) = no.u;
        }
        union { uint4 u; _Float16 h[8]; } tl;
        tl.h[0] = (_Float16)rx; tl.h[1] = (_Float16)ry;
        tl.h[2] = (_Float16)rz; tl.h[3] = (_Float16)dd;
        tl.h[4] = (_Float16)0.f; tl.h[5] = (_Float16)0.f;
        tl.h[6] = (_Float16)0.f; tl.h[7] = (_Float16)0.f;
        *reinterpret_cast<uint4*>(er + 128) = tl.u;
        const uint4 z = {0u, 0u, 0u, 0u};
        *reinterpret_cast<uint4*>(er + 136) = z;     // zero pad K = 136..159
        *reinterpret_cast<uint4*>(er + 144) = z;
        *reinterpret_cast<uint4*>(er + 152) = z;
    }
    __syncthreads();

    // ---- layer 1: [256 x 160] x [160 x 128] via v_wmma_f32_16x16x32_f16 ----
    // Two M-tiles in flight: the two independent accumulator chains fill each
    // other's WMMA->WMMA co-execution hazard slots (no v_nop bubbles).
    const int mtop = lane >> 4;                      // row-half select for D
    for (int mt = 0; mt < 16; mt += 2) {
        const _Float16* a0 = edge + (size_t)((mt    ) * 16 + (lane & 15)) * ES;
        const _Float16* a1 = edge + (size_t)((mt + 1) * 16 + (lane & 15)) * ES;
        v8f acc0 = {}, acc1 = {};
#pragma unroll
        for (int ks = 0; ks < 5; ++ks) {
            v16h x0 = load_tile16(a0 + ks * 32, lane);
            v16h x1 = load_tile16(a1 + ks * 32, lane);
            acc0 = __builtin_amdgcn_wmma_f32_16x16x32_f16(
                false, x0, false, B1[ks], (short)0, acc0, false, false);
            acc1 = __builtin_amdgcn_wmma_f32_16x16x32_f16(
                false, x1, false, B1[ks], (short)0, acc1, false, false);
        }
#pragma unroll
        for (int r = 0; r < 8; ++r) {                // D elem (m = r+8*mtop, n)
            const int m0 = (mt    ) * 16 + r + 8 * mtop;
            const int m1 = (mt + 1) * 16 + r + 8 * mtop;
            const float y0 = lrelu(fmaf(acc0[r], s1, o1));  // folded BN + LReLU
            const float y1 = lrelu(fmaf(acc1[r], s1, o1));
            h1[(size_t)m0 * H1S + crow] = (_Float16)y0;
            h1[(size_t)m1 * H1S + crow] = (_Float16)y1;
        }
    }
    __syncthreads();

    // ---- layer 2 + max over K + shortcut + final LReLU ----
    for (int mt = 0; mt < 16; mt += 2) {             // M-tile == one point
        const _Float16* a0 = h1 + (size_t)((mt    ) * 16 + (lane & 15)) * H1S;
        const _Float16* a1 = h1 + (size_t)((mt + 1) * 16 + (lane & 15)) * H1S;
        v8f acc0 = {}, acc1 = {};
#pragma unroll
        for (int ks = 0; ks < 4; ++ks) {
            v16h x0 = load_tile16(a0 + ks * 32, lane);
            v16h x1 = load_tile16(a1 + ks * 32, lane);
            acc0 = __builtin_amdgcn_wmma_f32_16x16x32_f16(
                false, x0, false, B2[ks], (short)0, acc0, false, false);
            acc1 = __builtin_amdgcn_wmma_f32_16x16x32_f16(
                false, x1, false, B2[ks], (short)0, acc1, false, false);
        }
        float mx0 = -3.4e38f, mx1 = -3.4e38f;
#pragma unroll
        for (int r = 0; r < 8; ++r) {
            mx0 = fmaxf(mx0, lrelu(fmaf(acc0[r], s2, o2)));
            mx1 = fmaxf(mx1, lrelu(fmaf(acc1[r], s2, o2)));
        }
        mx0 = fmaxf(mx0, __shfl_xor(mx0, 16, 32));   // combine k rows 0-7 / 8-15
        mx1 = fmaxf(mx1, __shfl_xor(mx1, 16, 32));
        if (lane < 16) {                             // lanes 16-31 duplicate
            const int p0 = pt0 + mt, p1 = pt0 + mt + 1;
            const float v0 = lrelu(mx0 + sc[((size_t)b * NPTS + p0) * COUT + crow]);
            const float v1 = lrelu(mx1 + sc[((size_t)b * NPTS + p1) * COUT + crow]);
            out[(size_t)b * COUT * NPTS + (size_t)crow * NPTS + p0] = v0;
            out[(size_t)b * COUT * NPTS + (size_t)crow * NPTS + p1] = v1;
        }
    }
}

// ---------------------------------------------------------------------------
extern "C" void kernel_launch(void* const* d_in, const int* in_sizes, int n_in,
                              void* d_out, int out_size, void* d_ws, size_t ws_size,
                              hipStream_t stream) {
    (void)in_sizes; (void)n_in; (void)out_size; (void)ws_size;
    const float* xyz   = (const float*)d_in[0];
    const float* feats = (const float*)d_in[1];
    const float* w_pre = (const float*)d_in[2];
    const float* bnp   = (const float*)d_in[3];
    const float* w_e1  = (const float*)d_in[4];
    const float* bne1  = (const float*)d_in[5];
    const float* w_e2  = (const float*)d_in[6];
    const float* bne2  = (const float*)d_in[7];
    const float* w_sc  = (const float*)d_in[8];
    const float* bnsc  = (const float*)d_in[9];
    float* out = (float*)d_out;

    char* ws = (char*)d_ws;
    size_t off = 0;
    auto alloc = [&](size_t bytes) -> void* {
        void* p = ws + off;
        off = (off + bytes + 255) & ~(size_t)255;
        return p;
    };
    int*      idx = (int*)     alloc((size_t)BATCH * NPTS * KNN  * sizeof(int));
    _Float16* fh  = (_Float16*)alloc((size_t)BATCH * NPTS * HPRE * sizeof(_Float16));
    float*    sc  = (float*)   alloc((size_t)BATCH * NPTS * COUT * sizeof(float));
    _Float16* w1h = (_Float16*)alloc((size_t)COUT * EINP * sizeof(_Float16));
    _Float16* w2h = (_Float16*)alloc((size_t)COUT * COUT * sizeof(_Float16));
    float*    e1s = (float*)   alloc(COUT * sizeof(float));
    float*    e1b = (float*)   alloc(COUT * sizeof(float));
    float*    e2s = (float*)   alloc(COUT * sizeof(float));
    float*    e2b = (float*)   alloc(COUT * sizeof(float));

    const size_t smem_bytes = (size_t)256 * ES * 2 + (size_t)256 * H1S * 2; // 152 KB
    (void)hipFuncSetAttribute((const void*)edge_kernel,
                              hipFuncAttributeMaxDynamicSharedMemorySize,
                              (int)smem_bytes);

    knn_kernel<<<dim3(NPTS / 256, BATCH), 256, 0, stream>>>(xyz, idx);
    prep_kernel<<<1, 256, 0, stream>>>(w_e1, bne1, w_e2, bne2,
                                       w1h, w2h, e1s, e1b, e2s, e2b);
    pre_kernel<<<dim3(NPTS / 4, BATCH), 256, 0, stream>>>(feats, w_pre, bnp,
                                                          w_sc, bnsc, fh, sc);
    edge_kernel<<<dim3(NPTS / TILE_PTS, BATCH), 256, smem_bytes, stream>>>(
        xyz, idx, fh, sc, w1h, w2h, e1s, e1b, e2s, e2b, out);
}
